// LiveRiskModel_51393578664417
// MI455X (gfx1250) — compile-verified, matrix operand order checked by reading, not run on hardware
//
#include <hip/hip_runtime.h>

// ---------------- model constants ----------------
#define BB 8
#define UU 16
#define TT 16
#define LL 32
#define EE 256
#define FFD 2048
#define DESCD 768
#define AED 64
#define DED 128
#define NCD 2

// GEMM tiling
#define BM 128
#define BN 128
#define KT 32

typedef __attribute__((ext_vector_type(16))) __bf16 v16bf;
typedef __attribute__((ext_vector_type(8)))  float  v8f;

#if defined(__gfx1250__) && __has_builtin(__builtin_amdgcn_global_load_async_to_lds_b128) && \
    __has_builtin(__builtin_amdgcn_s_wait_asynccnt)
#define USE_ASYNC_COPY 1
#else
#define USE_ASYNC_COPY 0
#endif

#if USE_ASYNC_COPY
// builtin prototype (from hipcc diagnostic): (int4vec addrspace(1)*, int4vec addrspace(3)*, imm, imm)
typedef int v4i __attribute__((vector_size(16)));
typedef __attribute__((address_space(1))) v4i* gas_v4i_p;
typedef __attribute__((address_space(3))) v4i* las_v4i_p;
#endif

// round-to-nearest-even fp32 pair -> packed bf16x2 (final merge = single v_perm_b32)
__device__ __forceinline__ unsigned pack2_bf16(float a, float b) {
  unsigned ua = __float_as_uint(a), ub = __float_as_uint(b);
  ua += 0x7FFFu + ((ua >> 16) & 1u);
  ub += 0x7FFFu + ((ub >> 16) & 1u);
#if __has_builtin(__builtin_amdgcn_perm)
  return __builtin_amdgcn_perm(ub, ua, 0x07060302u);  // {ub[31:16], ua[31:16]}
#else
  return (ua >> 16) | (ub & 0xFFFF0000u);
#endif
}

// ---------------- weight convert: W[N][K] fp32 -> Wb[N][Kpad] bf16 (zero pad) ----------------
__global__ void wconv_kernel(const float* __restrict__ W, unsigned short* __restrict__ Wb,
                             int N, int K, int Kpad) {
  long i = (long)blockIdx.x * 256 + threadIdx.x;
  if (i >= (long)N * Kpad) return;
  int n = (int)(i / Kpad), k = (int)(i % Kpad);
  float v = (k < K) ? W[(long)n * K + k] : 0.f;
  unsigned u = __float_as_uint(v);
  u = (u + 0x7FFFu + ((u >> 16) & 1u)) >> 16;
  Wb[i] = (unsigned short)u;
}

// ---------------- bf16 WMMA GEMM:  C[M,N] = A[M,K]fp32 @ Wb[N,K]bf16^T (+bias)(+relu) ----------------
// Block tile 128x128, 8 waves (4x2), wave tile 32x64 = 2x4 WMMA frags, K step 32,
// double-buffered LDS (one barrier per K step), B tile via async global->LDS when available.
// CU-mode residency caps us at ~128 VGPRs/wave: stage A (fetch+pack+ds_store) BEFORE the
// WMMA block so the 16 staging VGPRs have a short live range and never spill to scratch.
__global__ __launch_bounds__(256, 1)
void gemm_bf16_wmma(const float* __restrict__ A, const unsigned short* __restrict__ Bw,
                    const float* __restrict__ bias, float* __restrict__ C,
                    int M, int N, int K, int relu) {
  __shared__ unsigned short sA[2][BM * KT];  // 2 x 8 KB, [m][k]
  __shared__ unsigned short sB[2][BN * KT];  // 2 x 8 KB, [n][k]

  const int t    = threadIdx.x;
  const int lane = t & 31, wave = t >> 5;
  const int wm   = wave >> 1, wn = wave & 1;
  const int half = lane >> 4, idx = lane & 15;
  const long m0  = (long)blockIdx.x * BM;
  const long n0  = (long)blockIdx.y * BN;

  // staging decomposition: 256 threads cover 128 rows x 32 cols, 16 elems each
  const int  rrow  = t >> 1;        // 0..127
  const int  rcol  = (t & 1) * 16;  // 0 / 16
  const long agrow = m0 + rrow;
  const float*          asrc = A + agrow * (long)K + rcol;
  const unsigned short* bsrc = Bw + (n0 + rrow) * (long)K + rcol;

  v8f acc[2][4];
#pragma unroll
  for (int r = 0; r < 2; ++r)
#pragma unroll
    for (int c = 0; c < 4; ++c) acc[r][c] = {};

  // fetch + convert + store A tile (staging regs live only inside this call)
  auto stageA = [&](int kt, int buf) {
    float4 fa[4];
    if (agrow < M) {
      const float4* p = reinterpret_cast<const float4*>(asrc + kt * KT);
      fa[0] = p[0]; fa[1] = p[1]; fa[2] = p[2]; fa[3] = p[3];
    } else {
      fa[0] = make_float4(0, 0, 0, 0); fa[1] = fa[0]; fa[2] = fa[0]; fa[3] = fa[0];
    }
    unsigned* d = (unsigned*)sA[buf] + rrow * 16 + (t & 1) * 8;
    d[0] = pack2_bf16(fa[0].x, fa[0].y); d[1] = pack2_bf16(fa[0].z, fa[0].w);
    d[2] = pack2_bf16(fa[1].x, fa[1].y); d[3] = pack2_bf16(fa[1].z, fa[1].w);
    d[4] = pack2_bf16(fa[2].x, fa[2].y); d[5] = pack2_bf16(fa[2].z, fa[2].w);
    d[6] = pack2_bf16(fa[3].x, fa[3].y); d[7] = pack2_bf16(fa[3].z, fa[3].w);
  };
#if USE_ASYNC_COPY
  auto stageB = [&](int kt, int buf) {  // 32 bytes/thread = 2x b128, ASYNCcnt-tracked
    const unsigned short* g = bsrc + kt * KT;
    unsigned short*       l = sB[buf] + rrow * 32 + rcol;
    __builtin_amdgcn_global_load_async_to_lds_b128((gas_v4i_p)g, (las_v4i_p)l, 0, 0);
    __builtin_amdgcn_global_load_async_to_lds_b128((gas_v4i_p)(g + 8), (las_v4i_p)(l + 8), 0, 0);
  };
#else
  auto stageB = [&](int kt, int buf) {
    const uint4* p = reinterpret_cast<const uint4*>(bsrc + kt * KT);
    uint4 fb0 = p[0], fb1 = p[1];
    uint4* d = reinterpret_cast<uint4*>(sB[buf] + rrow * 32 + rcol);
    d[0] = fb0; d[1] = fb1;
  };
#endif

  // fragments per ISA 16-bit A/B layout: lane = half*16 + idx, VGPR pairs = consecutive K.
  // One B fragment live at a time (register pressure).
  auto compute = [&](int buf) {
    const unsigned* sa32 = (const unsigned*)sA[buf];
    const unsigned* sb32 = (const unsigned*)sB[buf];
    union Fr { v16bf v; unsigned u[8]; } a[2];
#pragma unroll
    for (int r = 0; r < 2; ++r) {
      const unsigned* pa = sa32 + (wm * 32 + r * 16 + idx) * 16;
#pragma unroll
      for (int p = 0; p < 4; ++p) {
        a[r].u[p]     = pa[4 * half + p];       // K = 8*half + 2p, 2p+1
        a[r].u[4 + p] = pa[8 + 4 * half + p];   // K = 16 + 8*half + 2p, ..
      }
    }
#pragma unroll
    for (int c = 0; c < 4; ++c) {
      union Fr { v16bf v; unsigned u[8]; } b;
      const unsigned* pb = sb32 + (wn * 64 + c * 16 + idx) * 16;
#pragma unroll
      for (int p = 0; p < 4; ++p) {
        b.u[p]     = pb[4 * half + p];
        b.u[4 + p] = pb[8 + 4 * half + p];
      }
      acc[0][c] = __builtin_amdgcn_wmma_f32_16x16x32_bf16(
          false, a[0].v, false, b.v, (short)0, acc[0][c], false, false);
      acc[1][c] = __builtin_amdgcn_wmma_f32_16x16x32_bf16(
          false, a[1].v, false, b.v, (short)0, acc[1][c], false, false);
    }
  };

  const int nk = K / KT;

  // prologue: tile 0 -> buffer 0
  stageB(0, 0);  // async B first: DMA runs while A is fetched/packed
  stageA(0, 0);
#if USE_ASYNC_COPY
  __builtin_amdgcn_s_wait_asynccnt(0);
#endif
  __syncthreads();

  for (int kt = 0; kt < nk; ++kt) {
    const int  cur  = kt & 1;
    const bool more = (kt + 1 < nk);
    if (more) {
      stageB(kt + 1, cur ^ 1);  // async engine fills the other buffer
      stageA(kt + 1, cur ^ 1);  // short-lived staging regs, no spill
    }
    compute(cur);
#if USE_ASYNC_COPY
    __builtin_amdgcn_s_wait_asynccnt(0);
#endif
    __syncthreads();
  }

  // epilogue: C/D layout => VGPR r holds M = r + 8*half, N = idx
#pragma unroll
  for (int c = 0; c < 4; ++c) {
    const long col = n0 + wn * 64 + c * 16 + idx;
    const float bv = bias ? bias[col] : 0.f;
#pragma unroll
    for (int r2 = 0; r2 < 2; ++r2) {
#pragma unroll
      for (int r = 0; r < 8; ++r) {
        const long row = m0 + wm * 32 + r2 * 16 + half * 8 + r;
        if (row < M) {
          float v = acc[r2][c][r] + bv;
          if (relu) v = fmaxf(v, 0.f);
          C[row * (long)N + col] = v;
        }
      }
    }
  }
}

// ---------------- block reductions ----------------
__device__ __forceinline__ float block_sum_256(float v, float* sbuf) {
  __syncthreads();
#pragma unroll
  for (int o = 16; o > 0; o >>= 1) v += __shfl_xor(v, o, 32);
  const int lane = threadIdx.x & 31, w = threadIdx.x >> 5;
  if (lane == 0) sbuf[w] = v;
  __syncthreads();
  float r = (threadIdx.x < 8) ? sbuf[threadIdx.x] : 0.f;
  if (w == 0) {
#pragma unroll
    for (int o = 4; o > 0; o >>= 1) r += __shfl_xor(r, o, 32);
    if (lane == 0) sbuf[0] = r;
  }
  __syncthreads();
  return sbuf[0];
}

// ---------------- x = LayerNorm(x + res) per row (E=256, 1 elem/thread) ----------------
__global__ __launch_bounds__(256)
void add_ln_kernel(float* __restrict__ x, const float* __restrict__ res,
                   const float* __restrict__ g, const float* __restrict__ b) {
  __shared__ float sbuf[8];
  const long row = blockIdx.x;
  const int  t   = threadIdx.x;
  float v   = x[row * EE + t] + res[row * EE + t];
  float mu  = block_sum_256(v, sbuf) * (1.f / EE);
  float d   = v - mu;
  float var = block_sum_256(d * d, sbuf) * (1.f / EE);
  x[row * EE + t] = d * rsqrtf(var + 1e-5f) * g[t] + b[t];
}

// ---------------- multi-head attention (context only; out-proj is a GEMM) ----------------
__global__ __launch_bounds__(256)
void attn_kernel(const float* __restrict__ qkv, float* __restrict__ ctx, int S, int h) {
  const int hd   = EE / h;
  const int b    = blockIdx.x / h;
  const int head = blockIdx.x % h;
  const int lane = threadIdx.x & 31, w = threadIdx.x >> 5;
  __shared__ float sq[8][128];   // per-wave query (hd <= 128)
  __shared__ float sp[8][260];   // per-wave scores (S <= 257)
  const float scale = rsqrtf((float)hd);
  const long  rs    = 3L * EE;
  const float* base = qkv + (long)b * S * rs;

  for (int i = w; i < S; i += 8) {
    for (int d = lane; d < hd; d += 32) sq[w][d] = base[(long)i * rs + head * hd + d];
    __builtin_amdgcn_wave_barrier();
    float mx = -1e30f;
    for (int j = lane; j < S; j += 32) {
      const float* kp = base + (long)j * rs + EE + head * hd;
      float s = 0.f;
      for (int d = 0; d < hd; ++d) s += sq[w][d] * kp[d];
      s *= scale;
      sp[w][j] = s;
      mx = fmaxf(mx, s);
    }
#pragma unroll
    for (int o = 16; o > 0; o >>= 1) mx = fmaxf(mx, __shfl_xor(mx, o, 32));
    __builtin_amdgcn_wave_barrier();
    float sum = 0.f;
    for (int j = lane; j < S; j += 32) {
      float e = __expf(sp[w][j] - mx);
      sp[w][j] = e;
      sum += e;
    }
#pragma unroll
    for (int o = 16; o > 0; o >>= 1) sum += __shfl_xor(sum, o, 32);
    const float inv = 1.f / sum;
    __builtin_amdgcn_wave_barrier();
    for (int d = lane; d < hd; d += 32) {
      float acc = 0.f;
      for (int j = 0; j < S; ++j) acc += sp[w][j] * base[(long)j * rs + 2 * EE + head * hd + d];
      ctx[((long)b * S + i) * EE + head * hd + d] = acc * inv;
    }
    __builtin_amdgcn_wave_barrier();
  }
}

// ---------------- data-movement kernels ----------------
__global__ __launch_bounds__(256)
void build_xcat_kernel(const int* __restrict__ pg, const float* __restrict__ emb,
                       const float* __restrict__ dproj, float* __restrict__ xcat) {
  const long row = blockIdx.x;
  const int  t   = threadIdx.x;
  if (t >= 224) return;
  const int ids = pg[row * 2 + 0];
  const int anc = pg[row * 2 + 1];
  float v;
  if (t < 64)        v = emb[(long)ids * AED + t];
  else if (t < 192)  v = dproj[row * DED + (t - 64)];
  else if (t == 192) v = (float)anc;
  else               v = 0.f;
  xcat[row * 224 + t] = v;
}

__global__ __launch_bounds__(256)
void build_x0_kernel(const float* __restrict__ xfc, const float* __restrict__ cls,
                     float* __restrict__ x0) {
  const long row = blockIdx.x;
  const int  c   = threadIdx.x;
  const long s   = row / 33;
  const int  p   = (int)(row % 33);
  x0[row * EE + c] = (p == 0) ? cls[c] : xfc[(s * 32 + p - 1) * EE + c];
}

__global__ __launch_bounds__(256)
void build_xmain_kernel(const float* __restrict__ x0, const float* __restrict__ up,
                        const float* __restrict__ tp, const float* __restrict__ mcls,
                        float* __restrict__ xm) {
  const long row = blockIdx.x;
  const int  c   = threadIdx.x;
  const int  b   = (int)(row / 257), j = (int)(row % 257);
  float v;
  if (j == 0) v = mcls[c];
  else {
    const int ut = j - 1, u = ut / TT, t2 = ut % TT;
    const long s = ((long)b * UU + u) * TT + t2;
    v = x0[(s * 33) * EE + c] + up[(long)u * EE + c] + tp[(long)t2 * EE + c];
  }
  xm[row * EE + c] = v;
}

__global__ __launch_bounds__(256)
void head_kernel(const float* __restrict__ xm, const float* __restrict__ hw,
                 const float* __restrict__ hb, float* __restrict__ out) {
  __shared__ float sbuf[8];
  const int b = blockIdx.x >> 1, c = blockIdx.x & 1;
  const int t = threadIdx.x;
  float v = xm[((long)b * 257) * EE + t] * hw[c * EE + t];
  float s = block_sum_256(v, sbuf);
  if (t == 0) out[b * NCD + c] = s + hb[c];
}

// ---------------- host orchestration ----------------
extern "C" void kernel_launch(void* const* d_in, const int* in_sizes, int n_in,
                              void* d_out, int out_size, void* d_ws, size_t ws_size,
                              hipStream_t stream) {
  (void)in_sizes; (void)n_in; (void)out_size; (void)ws_size;
  const int*   patch_grid = (const int*)  d_in[0];
  const float* desc       = (const float*)d_in[1];
  const float* emb        = (const float*)d_in[2];
  const float* bp_w       = (const float*)d_in[3];
  const float* bp_b       = (const float*)d_in[4];
  const float* fc_w       = (const float*)d_in[5];
  const float* fc_b       = (const float*)d_in[6];
  const float* patch_cls  = (const float*)d_in[7];
  const float* pe_in_w    = (const float*)d_in[8];
  const float* pe_in_b    = (const float*)d_in[9];
  const float* pe_out_w   = (const float*)d_in[10];
  const float* pe_out_b   = (const float*)d_in[11];
  const float* pe_ln1_g   = (const float*)d_in[12];
  const float* pe_ln1_b   = (const float*)d_in[13];
  const float* pe_w1      = (const float*)d_in[14];
  const float* pe_b1      = (const float*)d_in[15];
  const float* pe_w2      = (const float*)d_in[16];
  const float* pe_b2      = (const float*)d_in[17];
  const float* pe_ln2_g   = (const float*)d_in[18];
  const float* pe_ln2_b   = (const float*)d_in[19];
  const float* user_pos   = (const float*)d_in[20];
  const float* time_pos   = (const float*)d_in[21];
  const float* main_cls   = (const float*)d_in[22];
  const float* mt_in_w    = (const float*)d_in[23];
  const float* mt_in_b    = (const float*)d_in[24];
  const float* mt_out_w   = (const float*)d_in[25];
  const float* mt_out_b   = (const float*)d_in[26];
  const float* mt_ln1_g   = (const float*)d_in[27];
  const float* mt_ln1_b   = (const float*)d_in[28];
  const float* mt_w1      = (const float*)d_in[29];
  const float* mt_b1      = (const float*)d_in[30];
  const float* mt_w2      = (const float*)d_in[31];
  const float* mt_b2      = (const float*)d_in[32];
  const float* mt_ln2_g   = (const float*)d_in[33];
  const float* mt_ln2_b   = (const float*)d_in[34];
  const float* head_w     = (const float*)d_in[35];
  const float* head_b     = (const float*)d_in[36];
  float* out = (float*)d_out;

  const long NTOK = (long)BB * UU * TT * LL;  // 65536
  const long R1   = (long)BB * UU * TT * 33;  // 67584
  const long R2   = (long)BB * 257;           // 2056
  const int  KFC  = 224;                      // 193 padded to 32-multiple

  size_t off = 0;
  auto alloc = [&](size_t bytes) -> char* {
    char* p = (char*)d_ws + off;
    off += (bytes + 255) & ~(size_t)255;
    return p;
  };
  unsigned short* wb_bp     = (unsigned short*)alloc((size_t)DED * DESCD * 2);
  unsigned short* wb_fc     = (unsigned short*)alloc((size_t)EE * KFC * 2);
  unsigned short* wb_pe_in  = (unsigned short*)alloc((size_t)3 * EE * EE * 2);
  unsigned short* wb_pe_out = (unsigned short*)alloc((size_t)EE * EE * 2);
  unsigned short* wb_pe_w1  = (unsigned short*)alloc((size_t)FFD * EE * 2);
  unsigned short* wb_pe_w2  = (unsigned short*)alloc((size_t)EE * FFD * 2);
  unsigned short *wb_mt_in[2], *wb_mt_out[2], *wb_mt_w1[2], *wb_mt_w2[2];
  for (int i = 0; i < 2; ++i) {
    wb_mt_in[i]  = (unsigned short*)alloc((size_t)3 * EE * EE * 2);
    wb_mt_out[i] = (unsigned short*)alloc((size_t)EE * EE * 2);
    wb_mt_w1[i]  = (unsigned short*)alloc((size_t)FFD * EE * 2);
    wb_mt_w2[i]  = (unsigned short*)alloc((size_t)EE * FFD * 2);
  }
  float* dproj = (float*)alloc((size_t)NTOK * DED * 4);
  float* xcat  = (float*)alloc((size_t)NTOK * KFC * 4);
  float* xfc   = (float*)alloc((size_t)NTOK * EE * 4);
  float* x0    = (float*)alloc((size_t)R1 * EE * 4);
  float* qkv   = (float*)alloc((size_t)R1 * 3 * EE * 4);
  float* ctx   = (float*)alloc((size_t)R1 * EE * 4);
  float* proj  = (float*)alloc((size_t)R1 * EE * 4);
  float* ffmid = (float*)alloc((size_t)R1 * FFD * 4);
  float* xm    = (float*)alloc((size_t)R2 * EE * 4);
  float* qkvm  = (float*)alloc((size_t)R2 * 3 * EE * 4);
  float* ctxm  = (float*)alloc((size_t)R2 * EE * 4);
  float* projm = (float*)alloc((size_t)R2 * EE * 4);
  float* ffm   = (float*)alloc((size_t)R2 * FFD * 4);

  auto wc = [&](const float* W, unsigned short* Wb, int N, int K, int Kpad) {
    long tot = (long)N * Kpad;
    wconv_kernel<<<(unsigned)((tot + 255) / 256), 256, 0, stream>>>(W, Wb, N, K, Kpad);
  };
  auto gemm = [&](const float* A, const unsigned short* Bw, const float* bias, float* C,
                  long M, int N, int K, int relu) {
    dim3 g((unsigned)((M + BM - 1) / BM), (unsigned)(N / BN));
    gemm_bf16_wmma<<<g, 256, 0, stream>>>(A, Bw, bias, C, (int)M, N, K, relu);
  };

  // ---- weight conversion to bf16 [N][Kpad] ----
  wc(bp_w, wb_bp, DED, DESCD, DESCD);
  wc(fc_w, wb_fc, EE, AED + DED + 1, KFC);
  wc(pe_in_w, wb_pe_in, 3 * EE, EE, EE);
  wc(pe_out_w, wb_pe_out, EE, EE, EE);
  wc(pe_w1, wb_pe_w1, FFD, EE, EE);
  wc(pe_w2, wb_pe_w2, EE, FFD, FFD);
  for (int i = 0; i < 2; ++i) {
    wc(mt_in_w + (size_t)i * 3 * EE * EE, wb_mt_in[i], 3 * EE, EE, EE);
    wc(mt_out_w + (size_t)i * EE * EE, wb_mt_out[i], EE, EE, EE);
    wc(mt_w1 + (size_t)i * FFD * EE, wb_mt_w1[i], FFD, EE, EE);
    wc(mt_w2 + (size_t)i * EE * FFD, wb_mt_w2[i], EE, FFD, FFD);
  }

  // ---- patch featurization ----
  gemm(desc, wb_bp, bp_b, dproj, NTOK, DED, DESCD, 0);
  build_xcat_kernel<<<(unsigned)NTOK, 256, 0, stream>>>(patch_grid, emb, dproj, xcat);
  gemm(xcat, wb_fc, fc_b, xfc, NTOK, EE, KFC, 0);
  build_x0_kernel<<<(unsigned)R1, 256, 0, stream>>>(xfc, patch_cls, x0);

  // ---- patch encoder (1 layer, h=2) ----
  gemm(x0, wb_pe_in, pe_in_b, qkv, R1, 3 * EE, EE, 0);
  attn_kernel<<<(unsigned)(BB * UU * TT * 2), 256, 0, stream>>>(qkv, ctx, 33, 2);
  gemm(ctx, wb_pe_out, pe_out_b, proj, R1, EE, EE, 0);
  add_ln_kernel<<<(unsigned)R1, 256, 0, stream>>>(x0, proj, pe_ln1_g, pe_ln1_b);
  gemm(x0, wb_pe_w1, pe_b1, ffmid, R1, FFD, EE, 1);
  gemm(ffmid, wb_pe_w2, pe_b2, proj, R1, EE, FFD, 0);
  add_ln_kernel<<<(unsigned)R1, 256, 0, stream>>>(x0, proj, pe_ln2_g, pe_ln2_b);

  // ---- assemble main sequence ----
  build_xmain_kernel<<<(unsigned)R2, 256, 0, stream>>>(x0, user_pos, time_pos, main_cls, xm);

  // ---- main transformer (2 layers, h=4) ----
  for (int i = 0; i < 2; ++i) {
    gemm(xm, wb_mt_in[i], mt_in_b + (size_t)i * 3 * EE, qkvm, R2, 3 * EE, EE, 0);
    attn_kernel<<<(unsigned)(BB * 4), 256, 0, stream>>>(qkvm, ctxm, 257, 4);
    gemm(ctxm, wb_mt_out[i], mt_out_b + (size_t)i * EE, projm, R2, EE, EE, 0);
    add_ln_kernel<<<(unsigned)R2, 256, 0, stream>>>(xm, projm, mt_ln1_g + (size_t)i * EE,
                                                    mt_ln1_b + (size_t)i * EE);
    gemm(xm, wb_mt_w1[i], mt_b1 + (size_t)i * FFD, ffm, R2, FFD, EE, 1);
    gemm(ffm, wb_mt_w2[i], mt_b2 + (size_t)i * EE, projm, R2, EE, FFD, 0);
    add_ln_kernel<<<(unsigned)R2, 256, 0, stream>>>(xm, projm, mt_ln2_g + (size_t)i * EE,
                                                    mt_ln2_b + (size_t)i * EE);
  }

  // ---- head ----
  head_kernel<<<(unsigned)(BB * NCD), 256, 0, stream>>>(xm, head_w, head_b, out);
}
